// Model_43207370998119
// MI455X (gfx1250) — matrix operand order repaired and checked by reading, not compiled, not run on hardware
//
#include <hip/hip_runtime.h>
#include <hip/hip_bf16.h>

typedef unsigned int u32;
typedef unsigned short u16;
typedef __attribute__((ext_vector_type(16))) __bf16 v16bf;
typedef __attribute__((ext_vector_type(8)))  float  v8f;

// ---------------- problem constants ----------------
#define Bsz   128
#define Tlen  300
#define Hdim  400
#define Udim  64
#define CDIMc 54
#define Mmix  20
#define NG    1600          // 4*H
// packed-K layouts (multiples of 32 for WMMA k-steps)
#define K1    480           // [x(0..2), w(3..56), pad..63, h1(64..463), pad..479]
#define KT1   15
#define K2    896           // [x(0..2), h1(3..402), w(403..456), pad..479, h2(480..879), pad..895]
#define KT2   28
#define KTH   13            // 416/32 : K-tiles covering one H=400 region

// ---------------- workspace layout (bytes) ----------------
#define OFF_W1   0u                              // 100*15*1024
#define OFF_W2   (OFF_W1 + 100u*15u*1024u)       // 100*28*1024
#define OFF_WK   (OFF_W2 + 100u*28u*1024u)       // 2*13*1024
#define OFF_WO   (OFF_WK + 2u*13u*1024u)         // 8*13*1024
#define OFF_B1   (OFF_WO + 8u*13u*1024u)         // 1600 f32
#define OFF_B2   (OFF_B1 + 1600u*4u)             // 1600 f32
#define OFF_BK   (OFF_B2 + 1600u*4u)             // 32 f32
#define OFF_BO   (OFF_BK + 32u*4u)               // 128 f32
#define OFF_PART (OFF_BO + 128u*4u)              // 8 f32

// ---------------- LDS layout (bytes) ----------------
#define SM_A1   0                       // u16 [16][480]
#define SM_A2   (SM_A1 + 16*K1*2)       // u16 [16][896]
#define SM_HST  (SM_A2 + 16*K2*2)       // u16 [16][400]
#define SM_C1   (SM_HST + 16*400*2)     // f32 [16][400]
#define SM_C2   (SM_C1 + 16*400*4)      // f32 [16][400]
#define SM_KG   (SM_C2 + 16*400*4)      // f32 [16][32]
#define SM_AL   (SM_KG + 16*32*4)       // f32 [16][10]
#define SM_BE   (SM_AL + 640)           // f32 [16][10]
#define SM_KP   (SM_BE + 640)           // f32 [16][10]  (kappa state)
#define SM_PHI  (SM_KP + 640)           // f32 [16][64]
#define SM_WV   (SM_PHI + 16*64*4)      // f32 [16][54]
#define SM_OUT  (SM_WV + 16*54*4)       // f32 [16][128]
#define SM_RL   (SM_OUT + 16*128*4)     // f32 [16]
#define SMEM_TOTAL (SM_RL + 64)         // = 127808 bytes

// ---------------- helpers ----------------
__device__ __forceinline__ u16 f2bf(float f) {
  union { float f; u32 u; } x; x.f = f;
  u32 u = x.u;
  u += 0x7fffu + ((u >> 16) & 1u);      // round-to-nearest-even
  return (u16)(u >> 16);
}

__device__ __forceinline__ float sigf(float x) { return 1.f / (1.f + __expf(-x)); }

union Frag32 { v16bf v; uint4 q[2]; };

// A fragment 16x32 bf16 from LDS (row-major, rowStride in bf16 units).
// Lane l: row = l%16, half = l/16; dwords 0..3 hold K=half*8+0..7, dwords 4..7 hold K=16+half*8+0..7.
__device__ __forceinline__ v16bf load_a(const u16* base, int rowStride, int r, int hlf, int col0) {
  Frag32 u;
  const u16* p = base + r * rowStride + col0 + hlf * 8;
  u.q[0] = *reinterpret_cast<const uint4*>(p);
  u.q[1] = *reinterpret_cast<const uint4*>(p + 16);
  return u.v;
}

// B fragment: prepacked — tile = 1024 contiguous bytes, lane reads its own 32 bytes.
__device__ __forceinline__ v16bf load_b(const u32* wpk, int tile, int lane) {
  Frag32 u;
  const uint4* p = reinterpret_cast<const uint4*>(wpk + tile * 256 + lane * 8);
  u.q[0] = p[0];
  u.q[1] = p[1];
  return u.v;
}

__device__ __forceinline__ v8f wmma_bf16(v16bf a, v16bf b, v8f c) {
  return __builtin_amdgcn_wmma_f32_16x16x32_bf16(false, a, false, b, (short)0, c, false, false);
}

#define V8Z {0.f,0.f,0.f,0.f,0.f,0.f,0.f,0.f}

// ---------------- weight packing ----------------
// dst tile (nt,kt) at dword offset (nt*ktiles+kt)*256; lane l dword v holds bf16 pair
// (W[n][k0], W[n][k0+1]) with n = nt*16 + l%16, k0 = kt*32 + (l/16)*8 + (v>>2)*16 + (v&3)*2.
// packed col k maps: [0,widthA) -> srcA col k ; [offB, offB+widthB) -> srcB col k-offB ; else 0.
__global__ void pack_kernel(u32* __restrict__ dst,
                            const float* __restrict__ srcA, int ldA, int widthA,
                            const float* __restrict__ srcB, int ldB, int offB, int widthB,
                            int nUsed, int ktiles, int ntiles) {
  int id = blockIdx.x * blockDim.x + threadIdx.x;
  int total = ntiles * ktiles * 256;
  if (id >= total) return;
  int tile = id >> 8, rem = id & 255;
  int lane = rem >> 3, v = rem & 7;
  int nt = tile / ktiles, kt = tile % ktiles;
  int n  = nt * 16 + (lane & 15);
  int k0 = kt * 32 + (lane >> 4) * 8 + (v >> 2) * 16 + (v & 3) * 2;
  float f[2] = {0.f, 0.f};
  if (n < nUsed) {
    for (int j = 0; j < 2; ++j) {
      int k = k0 + j;
      if (k < widthA)                              f[j] = srcA[n * ldA + k];
      else if (srcB && k >= offB && k < offB + widthB) f[j] = srcB[n * ldB + (k - offB)];
    }
  }
  dst[id] = (u32)f2bf(f[0]) | ((u32)f2bf(f[1]) << 16);
}

__global__ void bias_kernel(float* b1, float* b2, float* bkc, float* boc,
                            const float* bih1, const float* bhh1,
                            const float* bih2, const float* bhh2,
                            const float* bk, const float* bo) {
  int id = blockIdx.x * blockDim.x + threadIdx.x;
  if (id < 1600)            b1[id] = bih1[id] + bhh1[id];
  else if (id < 3200)       b2[id - 1600] = bih2[id - 1600] + bhh2[id - 1600];
  else if (id < 3232)       { int i = id - 3200; bkc[i] = (i < 30) ? bk[i] : 0.f; }
  else if (id < 3360)       { int i = id - 3232; boc[i] = (i < 121) ? bo[i] : 0.f; }
}

// ---------------- fused recurrent kernel ----------------
__global__ void __launch_bounds__(512, 1)
rnn_kernel(const float* __restrict__ x, const float* __restrict__ y,
           const float* __restrict__ cvec,
           const u32* __restrict__ W1pk, const u32* __restrict__ W2pk,
           const u32* __restrict__ Wkpk, const u32* __restrict__ Wopk,
           const float* __restrict__ b1, const float* __restrict__ b2,
           const float* __restrict__ bkc, const float* __restrict__ boc,
           float* __restrict__ partials) {
  extern __shared__ char smem[];
  u16*   A1  = (u16*)(smem + SM_A1);
  u16*   A2  = (u16*)(smem + SM_A2);
  u16*   HST = (u16*)(smem + SM_HST);
  float* C1  = (float*)(smem + SM_C1);
  float* C2  = (float*)(smem + SM_C2);
  float* KG  = (float*)(smem + SM_KG);
  float* AL  = (float*)(smem + SM_AL);
  float* BE  = (float*)(smem + SM_BE);
  float* KP  = (float*)(smem + SM_KP);
  float* PHI = (float*)(smem + SM_PHI);
  float* WV  = (float*)(smem + SM_WV);
  float* OUTB= (float*)(smem + SM_OUT);
  float* RL  = (float*)(smem + SM_RL);

  const int tid  = threadIdx.x;
  const int wv_  = tid >> 5;
  const int lane = tid & 31;
  const int hlf  = lane >> 4;
  const int rl   = lane & 15;
  const int b0   = blockIdx.x * 16;

  for (int i = tid; i < SMEM_TOTAL / 4; i += 512) ((u32*)smem)[i] = 0u;
  __syncthreads();

  float lossAcc = 0.f;

  for (int t = 0; t < Tlen; ++t) {
    // ---- Phase A: x_t and w_prev into A1 (A2 x cols too) ----
    if (tid < 48) {
      int r = tid / 3, i = tid % 3;
      u16 bv = f2bf(x[((b0 + r) * Tlen + t) * 3 + i]);
      A1[r * K1 + i] = bv;
      A2[r * K2 + i] = bv;
    }
    for (int id = tid; id < 16 * CDIMc; id += 512) {
      int r = id / CDIMc, c = id % CDIMc;
      A1[r * K1 + 3 + c] = f2bf(WV[id]);
    }
    __syncthreads();

    // ---- LSTM1: gates = A1 @ [Wih1|Whh1]^T, fused activations in registers ----
    for (int q = wv_; q < 25; q += 16) {
      v8f ai = V8Z, af = V8Z, ag = V8Z, ao = V8Z;
      for (int kt = 0; kt < KT1; ++kt) {
        v16bf a  = load_a(A1, K1, rl, hlf, kt * 32);
        v16bf bi = load_b(W1pk, (q      ) * KT1 + kt, lane);
        v16bf bf = load_b(W1pk, (25 + q ) * KT1 + kt, lane);
        v16bf bg = load_b(W1pk, (50 + q ) * KT1 + kt, lane);
        v16bf bo_= load_b(W1pk, (75 + q ) * KT1 + kt, lane);
        ai = wmma_bf16(a, bi , ai);
        af = wmma_bf16(a, bf , af);
        ag = wmma_bf16(a, bg , ag);
        ao = wmma_bf16(a, bo_, ao);
      }
      int j = q * 16 + rl;
      float Bi = b1[j], Bf = b1[400 + j], Bg = b1[800 + j], Bo = b1[1200 + j];
#pragma unroll
      for (int v = 0; v < 8; ++v) {
        int row = v + hlf * 8;
        float iv = sigf(ai[v] + Bi);
        float fv = sigf(af[v] + Bf);
        float gv = tanhf(ag[v] + Bg);
        float ov = sigf(ao[v] + Bo);
        float cn = fv * C1[row * 400 + j] + iv * gv;
        float hn = ov * tanhf(cn);
        C1[row * 400 + j] = cn;
        u16 hb = f2bf(hn);
        HST[row * 400 + j] = hb;          // staged: A1 h-region still being read
        A2[row * K2 + 3 + j] = hb;        // A2 not read this phase
      }
    }
    __syncthreads();
    for (int id = tid; id < 6400; id += 512)
      A1[(id / 400) * K1 + 64 + (id % 400)] = HST[id];
    __syncthreads();

    // ---- attention: kg = h1 @ Wk^T + bk (2 N-tiles) ----
    if (wv_ < 2) {
      v8f acc = V8Z;
      for (int kt = 0; kt < KTH; ++kt) {
        v16bf a = load_a(A1, K1, rl, hlf, 64 + kt * 32);
        v16bf b = load_b(Wkpk, wv_ * KTH + kt, lane);
        acc = wmma_bf16(a, b, acc);
      }
      float bb = bkc[wv_ * 16 + rl];
#pragma unroll
      for (int v = 0; v < 8; ++v)
        KG[(v + hlf * 8) * 32 + wv_ * 16 + rl] = acc[v] + bb;
    }
    __syncthreads();

    if (tid < 160) {
      int r = tid / 10, k = tid % 10;
      AL[r * 10 + k] = __expf(KG[r * 32 + k]);
      BE[r * 10 + k] = __expf(KG[r * 32 + 10 + k]);
      KP[r * 10 + k] += __expf(KG[r * 32 + 20 + k]);
    }
    __syncthreads();
    for (int id = tid; id < 16 * Udim; id += 512) {
      int r = id >> 6;
      float u = (float)(id & 63);
      float s = 0.f;
#pragma unroll
      for (int k = 0; k < 10; ++k) {
        float d = KP[r * 10 + k] - u;
        s += AL[r * 10 + k] * __expf(-BE[r * 10 + k] * d * d);
      }
      PHI[id] = s;
    }
    __syncthreads();
    for (int id = tid; id < 16 * CDIMc; id += 512) {
      int r = id / CDIMc, c = id % CDIMc;
      const float* cp = cvec + (size_t)(b0 + r) * Udim * CDIMc + c;
      float s = 0.f;
      for (int u = 0; u < Udim; ++u) s += PHI[r * 64 + u] * cp[u * CDIMc];
      WV[id] = s;
      A2[r * K2 + 403 + c] = f2bf(s);
    }
    __syncthreads();

    // ---- LSTM2 ----
    for (int q = wv_; q < 25; q += 16) {
      v8f ai = V8Z, af = V8Z, ag = V8Z, ao = V8Z;
      for (int kt = 0; kt < KT2; ++kt) {
        v16bf a  = load_a(A2, K2, rl, hlf, kt * 32);
        v16bf bi = load_b(W2pk, (q      ) * KT2 + kt, lane);
        v16bf bf = load_b(W2pk, (25 + q ) * KT2 + kt, lane);
        v16bf bg = load_b(W2pk, (50 + q ) * KT2 + kt, lane);
        v16bf bo_= load_b(W2pk, (75 + q ) * KT2 + kt, lane);
        ai = wmma_bf16(a, bi , ai);
        af = wmma_bf16(a, bf , af);
        ag = wmma_bf16(a, bg , ag);
        ao = wmma_bf16(a, bo_, ao);
      }
      int j = q * 16 + rl;
      float Bi = b2[j], Bf = b2[400 + j], Bg = b2[800 + j], Bo = b2[1200 + j];
#pragma unroll
      for (int v = 0; v < 8; ++v) {
        int row = v + hlf * 8;
        float iv = sigf(ai[v] + Bi);
        float fv = sigf(af[v] + Bf);
        float gv = tanhf(ag[v] + Bg);
        float ov = sigf(ao[v] + Bo);
        float cn = fv * C2[row * 400 + j] + iv * gv;
        float hn = ov * tanhf(cn);
        C2[row * 400 + j] = cn;
        HST[row * 400 + j] = f2bf(hn);
      }
    }
    __syncthreads();
    for (int id = tid; id < 6400; id += 512)
      A2[(id / 400) * K2 + 480 + (id % 400)] = HST[id];
    __syncthreads();

    // ---- output projection: out = h2 @ Wo^T + bo (8 N-tiles) ----
    if (wv_ < 8) {
      v8f acc = V8Z;
      for (int kt = 0; kt < KTH; ++kt) {
        v16bf a = load_a(A2, K2, rl, hlf, 480 + kt * 32);
        v16bf b = load_b(Wopk, wv_ * KTH + kt, lane);
        acc = wmma_bf16(a, b, acc);
      }
      float bb = boc[wv_ * 16 + rl];
#pragma unroll
      for (int v = 0; v < 8; ++v)
        OUTB[(v + hlf * 8) * 128 + wv_ * 16 + rl] = acc[v] + bb;
    }
    __syncthreads();

    // ---- mixture-density loss, fused per step ----
    if (tid < 16) {
      const float* yp = y + ((size_t)(b0 + tid) * Tlen + t) * 3;
      float y1 = yp[0], y2 = yp[1], ye = yp[2];
      const float* o = OUTB + tid * 128;
      const float eps = 1e-20f;
      float eos = 1.f / (1.f + __expf(o[0]));  // sigmoid(-out0)
      lossAcc += -logf((eos + eps) * ye + (1.f - eos + eps) * (1.f - ye));
      float mx = -1e30f;
      for (int m = 0; m < Mmix; ++m) mx = fmaxf(mx, o[1 + m]);
      float Z = 0.f, S = 0.f;
      for (int m = 0; m < Mmix; ++m) {
        float e   = __expf(o[1 + m] - mx);
        float mu1 = o[21 + m], mu2 = o[41 + m];
        float s1  = __expf(o[61 + m]);
        float s2  = __expf(o[81 + m]);
        float rho = tanhf(o[101 + m]);
        float d1 = (y1 - mu1) / s1, d2 = (y2 - mu2) / s2;
        float omr2 = 1.f - rho * rho;
        float zz = d1 * d1 + d2 * d2 - 2.f * rho * d1 * d2;
        float Ng = __expf(-zz / (2.f * omr2)) / (6.2831853071795864f * s1 * s2 * sqrtf(omr2));
        Z += e;
        S += e * Ng;
      }
      lossAcc += -logf(S / Z + eps);
    }
    __syncthreads();
  }

  if (tid < 16) RL[tid] = lossAcc;
  __syncthreads();
  if (tid == 0) {
    float s = 0.f;
    for (int i = 0; i < 16; ++i) s += RL[i];
    partials[blockIdx.x] = s;
  }
}

__global__ void reduce_kernel(const float* __restrict__ partials, float* __restrict__ out) {
  if (blockIdx.x == 0 && threadIdx.x == 0) {
    float s = 0.f;
    for (int i = 0; i < 8; ++i) s += partials[i];
    out[0] = s / (float)(Bsz * Tlen);
  }
}

// ---------------- launch ----------------
extern "C" void kernel_launch(void* const* d_in, const int* in_sizes, int n_in,
                              void* d_out, int out_size, void* d_ws, size_t ws_size,
                              hipStream_t stream) {
  (void)in_sizes; (void)n_in; (void)out_size; (void)ws_size;
  const float* x    = (const float*)d_in[0];
  const float* y    = (const float*)d_in[1];
  const float* cvec = (const float*)d_in[2];
  const float* Wih1 = (const float*)d_in[3];
  const float* Whh1 = (const float*)d_in[4];
  const float* bih1 = (const float*)d_in[5];
  const float* bhh1 = (const float*)d_in[6];
  const float* Wih2 = (const float*)d_in[7];
  const float* Whh2 = (const float*)d_in[8];
  const float* bih2 = (const float*)d_in[9];
  const float* bhh2 = (const float*)d_in[10];
  const float* Wk   = (const float*)d_in[11];
  const float* bk   = (const float*)d_in[12];
  const float* Wo   = (const float*)d_in[13];
  const float* bo   = (const float*)d_in[14];

  char* ws = (char*)d_ws;
  u32* W1pk = (u32*)(ws + OFF_W1);
  u32* W2pk = (u32*)(ws + OFF_W2);
  u32* Wkpk = (u32*)(ws + OFF_WK);
  u32* Wopk = (u32*)(ws + OFF_WO);
  float* b1 = (float*)(ws + OFF_B1);
  float* b2 = (float*)(ws + OFF_B2);
  float* bkc = (float*)(ws + OFF_BK);
  float* boc = (float*)(ws + OFF_BO);
  float* partials = (float*)(ws + OFF_PART);

  // pack weights into WMMA B-fragment layout (bf16)
  pack_kernel<<<(100 * 15 * 256 + 255) / 256, 256, 0, stream>>>(
      W1pk, Wih1, 57, 57, Whh1, 400, 64, 400, 1600, 15, 100);
  pack_kernel<<<(100 * 28 * 256 + 255) / 256, 256, 0, stream>>>(
      W2pk, Wih2, 457, 457, Whh2, 400, 480, 400, 1600, 28, 100);
  pack_kernel<<<(2 * 13 * 256 + 255) / 256, 256, 0, stream>>>(
      Wkpk, Wk, 400, 400, nullptr, 0, 0, 0, 30, 13, 2);
  pack_kernel<<<(8 * 13 * 256 + 255) / 256, 256, 0, stream>>>(
      Wopk, Wo, 400, 400, nullptr, 0, 0, 0, 121, 13, 8);
  bias_kernel<<<(3360 + 255) / 256, 256, 0, stream>>>(
      b1, b2, bkc, boc, bih1, bhh1, bih2, bhh2, bk, bo);

  rnn_kernel<<<8, 512, SMEM_TOTAL, stream>>>(
      x, y, cvec, W1pk, W2pk, Wkpk, Wopk, b1, b2, bkc, boc, partials);

  reduce_kernel<<<1, 1, 0, stream>>>(partials, (float*)d_out);
}
